// QuantizingEncodecEncoder_19258633355397
// MI455X (gfx1250) — matrix operand order repaired and checked by reading, not compiled, not run on hardware
//
#include <hip/hip_runtime.h>
#include <hip/hip_bf16.h>
#include <math.h>
#include <stdint.h>

// ---------------------------------------------------------------------------
// Types for CDNA5 WMMA (wave32): v_wmma_f32_16x16x32_bf16
// ---------------------------------------------------------------------------
typedef __attribute__((ext_vector_type(16))) __bf16        v16bf;
typedef __attribute__((ext_vector_type(8)))  float         v8f;
typedef __attribute__((ext_vector_type(8)))  unsigned short us8;

union FragBF16 {
    v16bf v;
    us8   h[2];
};

__device__ __forceinline__ unsigned short f2bf(float f) {
    unsigned u = __float_as_uint(f);
    u += 0x7FFFu + ((u >> 16) & 1u);   // round-to-nearest-even
    return (unsigned short)(u >> 16);
}
__device__ __forceinline__ float eluf(float x) { return x > 0.f ? x : (__expf(x) - 1.f); }
__device__ __forceinline__ float sigf(float x) { return 1.f / (1.f + __expf(-x)); }

// ---------------------------------------------------------------------------
// Tensor Data Mover (TDM) support: async 2-D bf16 tile copy global -> LDS.
// Descriptor encoding per CDNA5 ISA §8.3/8.4; tracked by TENSORcnt.
// ---------------------------------------------------------------------------
#if __has_builtin(__builtin_amdgcn_tensor_load_to_lds) && \
    __has_builtin(__builtin_amdgcn_s_wait_tensorcnt)
#define HAVE_TDM 1
typedef unsigned int u32x4 __attribute__((ext_vector_type(4)));
typedef int          i32x4 __attribute__((ext_vector_type(4)));
typedef int          i32x8 __attribute__((ext_vector_type(8)));

// Copy a [rows x ck] tile of 2-byte elements (row stride = rowStrideElems in
// the source tensor) from global memory to LDS at byte offset lds_off.
__device__ __forceinline__ void tdm_load_2d_bf16(unsigned lds_off, const void* gptr,
                                                 int rows, int ck, int rowStrideElems) {
    unsigned long long ga = (unsigned long long)(uintptr_t)gptr;
    u32x4 g0;
    g0[0] = 1u;                                              // count=1, user mode
    g0[1] = lds_off;                                         // lds_addr (bytes)
    g0[2] = (unsigned)(ga & 0xFFFFFFFFu);                    // global_addr[31:0]
    g0[3] = (unsigned)((ga >> 32) & 0x1FFFFFFu) | (2u << 30); // addr[56:32] | type=2
    i32x8 g1;
    g1[0] = 0x10000;                         // data_size=1 (2B); mask=0; no pad/iter
    g1[1] = (int)((ck & 0xFFFF) << 16);      // tensor_dim0[15:0]
    g1[2] = (int)((rows & 0xFFFF) << 16);    // tensor_dim0 hi=0 | tensor_dim1[15:0]
    g1[3] = (int)((ck & 0xFFFF) << 16);      // tensor_dim1 hi=0 | tile_dim0
    g1[4] = rows;                            // tile_dim1 | tile_dim2=0
    g1[5] = rowStrideElems;                  // tensor_dim0_stride[31:0]
    g1[6] = 0;                               // stride0 hi | tensor_dim1_stride lo
    g1[7] = 0;
    i32x4 g2 = {};
    i32x4 g3 = {};
#if __clang_major__ >= 23
    i32x8 g4 = {};
    __builtin_amdgcn_tensor_load_to_lds(g0, g1, g2, g3, g4, 0);
#else
    __builtin_amdgcn_tensor_load_to_lds(g0, g1, g2, g3, 0);
#endif
}
#else
#define HAVE_TDM 0
#endif

// ---------------------------------------------------------------------------
// Direct 1-D convolution (memory-bound small-channel layers only).
// ---------------------------------------------------------------------------
__global__ void conv1d_direct(const float* __restrict__ in, const float* __restrict__ w,
                              const float* __restrict__ bias, const float* __restrict__ resid,
                              float* __restrict__ out,
                              int B, int Cin, int Lin, int Cout, int Lout,
                              int K, int stride, int pl, int elu_in) {
    long long idx   = (long long)blockIdx.x * blockDim.x + threadIdx.x;
    long long total = (long long)B * Cout * Lout;
    if (idx >= total) return;
    int lo = (int)(idx % Lout);
    int co = (int)((idx / Lout) % Cout);
    int b  = (int)(idx / ((long long)Lout * Cout));
    float sum = bias[co];
    const float* wp = w + (long long)co * Cin * K;
    const float* ip = in + (long long)b * Cin * Lin;
    int li0 = lo * stride - pl;
    for (int ci = 0; ci < Cin; ++ci) {
        for (int k = 0; k < K; ++k) {
            int li = li0 + k;
            if (li >= 0 && li < Lin) {
                float v = ip[(long long)ci * Lin + li];
                if (elu_in) v = eluf(v);
                sum += v * wp[ci * K + k];
            }
        }
    }
    if (resid) sum += resid[idx];
    out[idx] = sum;
}

// ---------------------------------------------------------------------------
// bf16 WMMA GEMM: C[M,N] = A[M,K] @ W[N,K]^T (+ bias[n]).
// Block = 64 threads = 2 waves; each wave owns 4 consecutive 16x16 tiles.
// With TDM: the 16xK A panel is DMA'd into LDS in 512-col chunks, double-
// buffered so the next chunk's tensor_load_to_lds overlaps current WMMAs.
// Requires M%16==0, K%32==0, N%64==0 (all call sites satisfy this).
// ---------------------------------------------------------------------------
__global__ void gemm_bf16_wmma(const unsigned short* __restrict__ A,
                               const unsigned short* __restrict__ W,
                               const float* __restrict__ bias,
                               float* __restrict__ C,
                               int M, int N, int K) {
    int lane = threadIdx.x & 31;
    int wave = threadIdx.x >> 5;                 // 0..1
    int m0 = blockIdx.x * 16;
    int n0 = blockIdx.y * 128 + wave * 64;       // this wave: cols n0..n0+63
    bool active = (n0 < N);                      // wave-uniform
    int akb = (lane >> 4) << 3;                  // 0 or 8
    int bkb = (lane >> 4) << 3;                  // 0 or 8
    const unsigned short* bp0 = W + (size_t)((active ? n0 : 0) + (lane & 15)) * K + bkb;
    v8f acc[4];
    for (int t = 0; t < 4; ++t) acc[t] = (v8f){};

#if HAVE_TDM
    extern __shared__ __align__(16) unsigned short As[];   // 2 * 16 * 512
    const int CH = 512;
    {
        int ck0 = K < CH ? K : CH;
        if (wave == 0) {
            tdm_load_2d_bf16((unsigned)(uintptr_t)&As[0], A + (size_t)m0 * K, 16, ck0, K);
            __builtin_amdgcn_s_wait_tensorcnt(0);
        }
    }
    __syncthreads();
    int buf = 0;
    for (int kc = 0; kc < K; kc += CH) {
        int ck   = (K - kc) < CH ? (K - kc) : CH;
        int kn   = kc + CH;
        bool more = kn < K;
        if (wave == 0 && more) {
            int ckn = (K - kn) < CH ? (K - kn) : CH;
            tdm_load_2d_bf16((unsigned)(uintptr_t)&As[(buf ^ 1) * 16 * CH],
                             A + (size_t)m0 * K + kn, 16, ckn, K);
        }
        if (active) {
            const unsigned short* as = &As[buf * 16 * CH + (lane & 15) * ck + akb];
            for (int kl = 0; kl < ck; kl += 32) {
                FragBF16 a;
                a.h[0] = *(const us8*)(as + kl);
                a.h[1] = *(const us8*)(as + kl + 16);
                for (int t = 0; t < 4; ++t) {
                    const unsigned short* bp = bp0 + (size_t)t * 16 * K + kc;
                    FragBF16 b;
                    b.h[0] = *(const us8*)(bp + kl);
                    b.h[1] = *(const us8*)(bp + kl + 16);
                    acc[t] = __builtin_amdgcn_wmma_f32_16x16x32_bf16(false, a.v, false, b.v,
                                                                     (short)0, acc[t], false, false);
                }
            }
        }
        if (wave == 0 && more) __builtin_amdgcn_s_wait_tensorcnt(0);
        __syncthreads();
        buf ^= 1;
    }
#else
    if (active) {
        const unsigned short* ap = A + (size_t)(m0 + (lane & 15)) * K + akb;
        for (int k0 = 0; k0 < K; k0 += 32) {
            __builtin_prefetch(ap + k0 + 64, 0, 1);
            FragBF16 a;
            a.h[0] = *(const us8*)(ap + k0);
            a.h[1] = *(const us8*)(ap + k0 + 16);
            for (int t = 0; t < 4; ++t) {
                const unsigned short* bp = bp0 + (size_t)t * 16 * K;
                FragBF16 b;
                b.h[0] = *(const us8*)(bp + k0);
                b.h[1] = *(const us8*)(bp + k0 + 16);
                acc[t] = __builtin_amdgcn_wmma_f32_16x16x32_bf16(false, a.v, false, b.v,
                                                                 (short)0, acc[t], false, false);
            }
        }
    }
#endif

    if (active) {
        int mrow = m0 + (lane >> 4) * 8;
        int ncb  = n0 + (lane & 15);
        for (int t = 0; t < 4; ++t) {
            int ncol = ncb + t * 16;
            float bv = bias ? bias[ncol] : 0.f;
            for (int j = 0; j < 8; ++j)
                C[(size_t)(mrow + j) * N + ncol] = acc[t][j] + bv;
        }
    }
}

// ---------------------------------------------------------------------------
// im2col for downsample convs: [B,Cin,Lin] f32 -> [B*Lout, Cin*Kw] bf16,
// ELU fused. Column order ci*Kw+kw matches [Cout, Cin, Kw] weight rows.
// ---------------------------------------------------------------------------
__global__ void im2col_down(const float* __restrict__ in, unsigned short* __restrict__ col,
                            int B, int Cin, int Lin, int Lout, int Kw, int stride, int pl) {
    long long idx   = (long long)blockIdx.x * blockDim.x + threadIdx.x;
    long long total = (long long)B * Lout * Cin * Kw;
    if (idx >= total) return;
    int kw = (int)(idx % Kw);
    int ci = (int)((idx / Kw) % Cin);
    long long nl = idx / ((long long)Kw * Cin);
    int lo = (int)(nl % Lout);
    int b  = (int)(nl / Lout);
    int li = lo * stride + kw - pl;
    float v = 0.f;
    if (li >= 0 && li < Lin) v = eluf(in[((long long)b * Cin + ci) * Lin + li]);
    col[idx] = f2bf(v);
}

__global__ void convert_bf16(const float* __restrict__ src, unsigned short* __restrict__ dst,
                             long long n) {
    long long idx = (long long)blockIdx.x * blockDim.x + threadIdx.x;
    if (idx >= n) return;
    dst[idx] = f2bf(src[idx]);
}

// [B*Lout, Cout] f32 -> [B, Cout, Lout] f32
__global__ void nlc_to_ncl(const float* __restrict__ in, float* __restrict__ out,
                           int B, int Cout, int Lout) {
    long long idx   = (long long)blockIdx.x * blockDim.x + threadIdx.x;
    long long total = (long long)B * Cout * Lout;
    if (idx >= total) return;
    int lo = (int)(idx % Lout);
    int co = (int)((idx / Lout) % Cout);
    int b  = (int)(idx / ((long long)Lout * Cout));
    out[idx] = in[((long long)b * Lout + lo) * Cout + co];
}

// ---------------------------------------------------------------------------
// LSTM weight repack: permute gate rows so hidden unit j's 4 gates land in
// 4 consecutive 16-column WMMA tiles: n' = (j/16)*64 + gate*16 + (j%16).
// ---------------------------------------------------------------------------
__global__ void lstm_repack(const float* __restrict__ wih, const float* __restrict__ whh,
                            const float* __restrict__ bih, const float* __restrict__ bhh,
                            unsigned short* __restrict__ wihp, unsigned short* __restrict__ whhp,
                            float* __restrict__ biasp) {
    int idx = blockIdx.x * blockDim.x + threadIdx.x;   // 2048*512
    if (idx >= 2048 * 512) return;
    int k    = idx & 511;
    int np   = idx >> 9;
    int u    = np >> 6;
    int wi   = np & 63;
    int gate = wi >> 4;
    int j    = u * 16 + (wi & 15);
    int orig = gate * 512 + j;
    wihp[idx] = f2bf(wih[(size_t)orig * 512 + k]);
    whhp[idx] = f2bf(whh[(size_t)orig * 512 + k]);
    if (k == 0) biasp[np] = bih[orig] + bhh[orig];
}

// [B=16, 512, T=225] -> row-major [T*16, 512] (f32 + bf16 bits)
__global__ void transpose_enc(const float* __restrict__ enc, float* __restrict__ yf,
                              unsigned short* __restrict__ yb) {
    int idx = blockIdx.x * blockDim.x + threadIdx.x;   // 16*512*225
    if (idx >= 16 * 512 * 225) return;
    int t = idx % 225;
    int c = (idx / 225) % 512;
    int b = idx / (225 * 512);
    float v = enc[idx];
    size_t o = (size_t)(t * 16 + b) * 512 + c;
    yf[o] = v;
    yb[o] = f2bf(v);
}

// ---------------------------------------------------------------------------
// Persistent LSTM recurrence: one workgroup, 32 waves, T sequential steps.
// h in LDS (bf16, 16x512); c in registers (layout == WMMA C fragment).
// Wave u owns hidden units [16u,16u+16): its 4 WMMA tiles = gates i,f,g,o.
// ---------------------------------------------------------------------------
__global__ void __launch_bounds__(1024)
lstm_recurrent(const float* __restrict__ Xp,            // [T*16, 2048] permuted x-proj
               const unsigned short* __restrict__ Whhp, // [2048, 512] permuted bf16
               const float* __restrict__ biasp,         // [2048] permuted
               unsigned short* __restrict__ hseq,       // [T*16, 512] bf16 out
               float* __restrict__ zf32,                // optional f32 out
               int T) {
    __shared__ __align__(16) unsigned short hbuf[16 * 512];
    int lane  = threadIdx.x & 31;
    int u     = threadIdx.x >> 5;      // 0..31
    int jn    = lane & 15;
    int hi    = lane >> 4;
    int akb   = hi * 8;
    int bkb   = hi * 8;
    int abrow = lane & 15;             // batch row for A fragment

    for (int i = threadIdx.x; i < 16 * 512; i += 1024) hbuf[i] = 0;
    float creg[8];
    for (int j = 0; j < 8; ++j) creg[j] = 0.f;

    int   col[4];
    float bgate[4];
    const unsigned short* bp[4];
    for (int g = 0; g < 4; ++g) {
        col[g]   = u * 64 + g * 16 + jn;
        bgate[g] = biasp[col[g]];
        bp[g]    = Whhp + (size_t)col[g] * 512 + bkb;
    }
    __syncthreads();

    for (int t = 0; t < T; ++t) {
        v8f acc[4];
        for (int g = 0; g < 4; ++g)
            for (int j = 0; j < 8; ++j)
                acc[g][j] = Xp[(size_t)(t * 16 + j + hi * 8) * 2048 + col[g]] + bgate[g];

        for (int k0 = 0; k0 < 512; k0 += 32) {
            FragBF16 a;
            a.h[0] = *(const us8*)&hbuf[abrow * 512 + k0 + akb];
            a.h[1] = *(const us8*)&hbuf[abrow * 512 + k0 + akb + 16];
            for (int g = 0; g < 4; ++g) {
                FragBF16 b;
                b.h[0] = *(const us8*)(bp[g] + k0);
                b.h[1] = *(const us8*)(bp[g] + k0 + 16);
                acc[g] = __builtin_amdgcn_wmma_f32_16x16x32_bf16(false, a.v, false, b.v,
                                                                 (short)0, acc[g], false, false);
            }
        }
        __syncthreads();   // all waves finished reading old h
        for (int j = 0; j < 8; ++j) {
            float iv = sigf(acc[0][j]);
            float fv = sigf(acc[1][j]);
            float gv = tanhf(acc[2][j]);
            float ov = sigf(acc[3][j]);
            float c  = fv * creg[j] + iv * gv;
            creg[j]  = c;
            float hv = ov * tanhf(c);
            int m = j + hi * 8;
            unsigned short hb = f2bf(hv);
            hbuf[m * 512 + u * 16 + jn] = hb;
            size_t gidx = (size_t)(t * 16 + m) * 512 + u * 16 + jn;
            hseq[gidx] = hb;
            if (zf32) zf32[gidx] = hv;
        }
        __syncthreads();   // new h visible before next step
    }
}

// ---------------------------------------------------------------------------
// conv_out as GEMM: im2col of elu(y+z) with k=7 taps (pad 3), tap-major cols.
// ---------------------------------------------------------------------------
__global__ void im2col_convout(const float* __restrict__ y, const float* __restrict__ z,
                               unsigned short* __restrict__ colbuf) {
    long long idx = (long long)blockIdx.x * blockDim.x + threadIdx.x;   // 3600*3584
    if (idx >= 3600LL * 3584) return;
    int cidx = (int)(idx % 3584);
    int n    = (int)(idx / 3584);
    int dt = cidx / 512, c = cidx % 512;
    int t = n >> 4, m = n & 15;
    int ts = t + dt - 3;
    float v = 0.f;
    if (ts >= 0 && ts < 225) {
        size_t s = (size_t)(ts * 16 + m) * 512 + c;
        v = eluf(y[s] + z[s]);
    }
    colbuf[idx] = f2bf(v);
}

__global__ void pack_wout(const float* __restrict__ w, unsigned short* __restrict__ wp) {
    int idx = blockIdx.x * blockDim.x + threadIdx.x;   // 128*3584
    if (idx >= 128 * 3584) return;
    int cidx = idx % 3584, co = idx / 3584;
    int dt = cidx / 512, c = cidx % 512;
    wp[idx] = f2bf(w[((size_t)co * 512 + c) * 7 + dt]);
}

// ---------------------------------------------------------------------------
// RVQ helpers
// ---------------------------------------------------------------------------
__global__ void rvq_prep(const float* __restrict__ cb, unsigned short* __restrict__ cbb,
                         float* __restrict__ cb2) {
    int idx = blockIdx.x * blockDim.x + threadIdx.x;   // 8*1024
    if (idx >= 8192) return;
    const float* p = cb + (size_t)idx * 128;
    float s = 0.f;
    for (int d = 0; d < 128; ++d) { float v = p[d]; s += v * v; cbb[(size_t)idx * 128 + d] = f2bf(v); }
    cb2[idx] = s;
}

__global__ void rvq_init(const float* __restrict__ hpre, float* __restrict__ res,
                         unsigned short* __restrict__ resb, float* __restrict__ quant) {
    int idx = blockIdx.x * blockDim.x + threadIdx.x;   // 3600*128
    if (idx >= 3600 * 128) return;
    float v = hpre[idx];
    res[idx]   = v;
    resb[idx]  = f2bf(v);
    quant[idx] = 0.f;
}

__global__ void rvq_argmin_update(const float* __restrict__ scores, const float* __restrict__ cb2all,
                                  const float* __restrict__ cball, float* __restrict__ res,
                                  unsigned short* __restrict__ resb, float* __restrict__ quant,
                                  int stage) {
    __shared__ float sv[128];
    __shared__ int   si[128];
    int row = blockIdx.x, tid = threadIdx.x;
    const float* c2 = cb2all + stage * 1024;
    const float* sr = scores + (size_t)row * 1024;
    float best = 3.4e38f; int bi = 0;
    for (int n = tid; n < 1024; n += 128) {
        float d = c2[n] - 2.f * sr[n];
        if (d < best) { best = d; bi = n; }
    }
    sv[tid] = best; si[tid] = bi;
    __syncthreads();
    for (int s = 64; s > 0; s >>= 1) {
        if (tid < s) {
            float ov = sv[tid + s]; int oi = si[tid + s];
            if (ov < sv[tid] || (ov == sv[tid] && oi < si[tid])) { sv[tid] = ov; si[tid] = oi; }
        }
        __syncthreads();
    }
    int idx = si[0];
    float qv = cball[((size_t)stage * 1024 + idx) * 128 + tid];
    size_t o = (size_t)row * 128 + tid;
    float r = res[o] - qv;
    res[o]  = r;
    resb[o] = f2bf(r);
    quant[o] += qv;
}

__global__ void final_write(const float* __restrict__ quant, float* __restrict__ out) {
    int idx = blockIdx.x * blockDim.x + threadIdx.x;   // 16*128*225
    if (idx >= 16 * 128 * 225) return;
    int t = idx % 225;
    int d = (idx / 225) % 128;
    int b = idx / (225 * 128);
    out[idx] = quant[(size_t)(t * 16 + b) * 128 + d];
}

// ---------------------------------------------------------------------------
// Host orchestration
// ---------------------------------------------------------------------------
extern "C" void kernel_launch(void* const* d_in, const int* in_sizes, int n_in,
                              void* d_out, int out_size, void* d_ws, size_t ws_size,
                              hipStream_t stream) {
    (void)in_sizes; (void)n_in; (void)out_size; (void)ws_size;
    auto F = [&](int i) { return (const float*)d_in[i]; };
    const float* x = F(0);

    char* base = (char*)d_ws;
    size_t off = 0;
    auto alloc = [&](size_t bytes) -> char* {
        off = (off + 255) & ~(size_t)255;
        char* p = base + off;
        off += bytes;
        return p;
    };

    const size_t F0 = 36864000;                                // max [16,64,36000]
    float* bufP = (float*)alloc(F0 * 4);                       // h (NCL)
    float* bufR = (float*)alloc(F0 * 4);                       // tmp2 / im2col (as ushort)
    float* bufT = (float*)alloc(F0 * 4);                       // h2
    float* bufU = (float*)alloc(F0 * 4);                       // down GEMM out (NLC)
    float* bufQ = (float*)alloc((F0 / 2) * 4);                 // tmp1 (C/2)
    unsigned short* wdownb = (unsigned short*)alloc((size_t)512 * 4096 * 2);

    auto conv = [&](const float* in, const float* w, const float* b, const float* resid,
                    float* out, int B, int Cin, int Lin, int Cout, int Lout,
                    int K, int stride, int pl, int elu) {
        long long total = (long long)B * Cout * Lout;
        int blocks = (int)((total + 255) / 256);
        conv1d_direct<<<blocks, 256, 0, stream>>>(in, w, b, resid, out,
                                                  B, Cin, Lin, Cout, Lout, K, stride, pl, elu);
    };
    auto gemm = [&](const unsigned short* A, const unsigned short* W, const float* bias,
                    float* C, int M, int N, int K) {
        dim3 g(M / 16, (N + 127) / 128);
        gemm_bf16_wmma<<<g, 64, 2 * 16 * 512 * 2 /* LDS A double-buffer */, stream>>>(
            A, W, bias, C, M, N, K);
    };

    // ---- SEANet encoder front-end ----
    conv(x, F(1), F(2), nullptr, bufP, 16, 1, 72000, 32, 72000, 7, 1, 3, 0);
    int C = 32, L = 72000;
    const int ratios[4] = {2, 4, 5, 8};
    for (int i = 0; i < 4; ++i) {
        int pb = 3 + i * 8;
        int r  = ratios[i];
        // residual unit (direct conv; memory bound)
        conv(bufP, F(pb + 0), F(pb + 1), nullptr, bufQ, 16, C,     L, C / 2, L, 3, 1, 1, 1);
        conv(bufQ, F(pb + 2), F(pb + 3), nullptr, bufR, 16, C / 2, L, C,     L, 1, 1, 0, 1);
        conv(bufP, F(pb + 4), F(pb + 5), bufR,    bufT, 16, C,     L, C,     L, 1, 1, 0, 0);
        // downsample as im2col + WMMA GEMM: k=2r, stride=r, pad_total=r, pl=r-r/2
        int Lo = L / r;
        int Kw = 2 * r, pl = r - r / 2;
        int Kred = C * Kw;                              // 128/512/1280/4096, all %32==0
        long long icN = (long long)16 * Lo * Kred;
        unsigned short* icol = (unsigned short*)bufR;   // tmp2 already consumed
        im2col_down<<<(int)((icN + 255) / 256), 256, 0, stream>>>(bufT, icol,
                                                                  16, C, L, Lo, Kw, r, pl);
        long long wn = (long long)(2 * C) * Kred;
        convert_bf16<<<(int)((wn + 255) / 256), 256, 0, stream>>>(F(pb + 6), wdownb, wn);
        gemm(icol, wdownb, F(pb + 7), bufU, 16 * Lo, 2 * C, Kred);
        long long tn = (long long)16 * (2 * C) * Lo;
        nlc_to_ncl<<<(int)((tn + 255) / 256), 256, 0, stream>>>(bufU, bufP, 16, 2 * C, Lo);
        C *= 2; L = Lo;
    }
    // bufP now holds encoder output [16, 512, 225]

    // ---- LSTM bottleneck (WMMA) ----
    const int NB = 3600;   // T*B = 225*16
    float*          yf    = (float*)alloc((size_t)NB * 512 * 4);
    unsigned short* yb    = (unsigned short*)alloc((size_t)NB * 512 * 2);
    unsigned short* wihp0 = (unsigned short*)alloc(2048 * 512 * 2);
    unsigned short* whhp0 = (unsigned short*)alloc(2048 * 512 * 2);
    unsigned short* wihp1 = (unsigned short*)alloc(2048 * 512 * 2);
    unsigned short* whhp1 = (unsigned short*)alloc(2048 * 512 * 2);
    float*          bp0   = (float*)alloc(2048 * 4);
    float*          bp1   = (float*)alloc(2048 * 4);
    float*          Xp    = (float*)alloc((size_t)NB * 2048 * 4);
    unsigned short* h0s   = (unsigned short*)alloc((size_t)NB * 512 * 2);
    float*          zf    = (float*)alloc((size_t)NB * 512 * 4);

    transpose_enc<<<(16 * 512 * 225 + 255) / 256, 256, 0, stream>>>(bufP, yf, yb);
    lstm_repack<<<4096, 256, 0, stream>>>(F(35), F(36), F(37), F(38), wihp0, whhp0, bp0);
    lstm_repack<<<4096, 256, 0, stream>>>(F(39), F(40), F(41), F(42), wihp1, whhp1, bp1);

    gemm(yb, wihp0, nullptr, Xp, NB, 2048, 512);
    lstm_recurrent<<<1, 1024, 0, stream>>>(Xp, whhp0, bp0, h0s, nullptr, 225);
    gemm(h0s, wihp1, nullptr, Xp, NB, 2048, 512);
    lstm_recurrent<<<1, 1024, 0, stream>>>(Xp, whhp1, bp1, h0s, zf, 225);

    // ---- conv_out as im2col GEMM (input = elu(y + z)) ----
    unsigned short* colbuf = (unsigned short*)alloc((size_t)NB * 3584 * 2);
    unsigned short* woutp  = (unsigned short*)alloc((size_t)128 * 3584 * 2);
    float*          hpre   = (float*)alloc((size_t)NB * 128 * 4);
    pack_wout<<<(128 * 3584 + 255) / 256, 256, 0, stream>>>(F(43), woutp);
    im2col_convout<<<(int)((3600LL * 3584 + 255) / 256), 256, 0, stream>>>(yf, zf, colbuf);
    gemm(colbuf, woutp, F(44), hpre, NB, 128, 3584);

    // ---- Residual VQ (8 stages, GEMM scores + argmin update) ----
    float*          res    = (float*)alloc((size_t)NB * 128 * 4);
    unsigned short* resb   = (unsigned short*)alloc((size_t)NB * 128 * 2);
    float*          quant  = (float*)alloc((size_t)NB * 128 * 4);
    unsigned short* cbb    = (unsigned short*)alloc((size_t)8 * 1024 * 128 * 2);
    float*          cb2    = (float*)alloc(8192 * 4);
    float*          scores = (float*)alloc((size_t)NB * 1024 * 4);

    rvq_prep<<<32, 256, 0, stream>>>(F(45), cbb, cb2);
    rvq_init<<<(NB * 128 + 255) / 256, 256, 0, stream>>>(hpre, res, resb, quant);
    for (int q = 0; q < 8; ++q) {
        gemm(resb, cbb + (size_t)q * 1024 * 128, nullptr, scores, NB, 1024, 128);
        rvq_argmin_update<<<NB, 128, 0, stream>>>(scores, cb2, F(45), res, resb, quant, q);
    }

    final_write<<<(16 * 128 * 225 + 255) / 256, 256, 0, stream>>>(quant, (float*)d_out);
}
